// graph_emb_67276367724817
// MI455X (gfx1250) — matrix operand order repaired and checked by reading, not compiled, not run on hardware
//
#include <hip/hip_runtime.h>

#define N_NODES 50000
#define D_MODEL 768
#define N_EDGES 100000
#define M_PAD   50048   /* 128 * 391 */

typedef __attribute__((ext_vector_type(16))) __bf16          v16bf;
typedef __attribute__((ext_vector_type(8)))  float           v8f;
typedef __attribute__((ext_vector_type(8)))  unsigned short  ushort8;

union BfFrag { ushort8 h[2]; v16bf v; };

__device__ __forceinline__ unsigned short f2bf(float f) {
    // round-to-nearest-even fp32 -> bf16
    unsigned int u = __float_as_uint(f);
    u += 0x7FFFu + ((u >> 16) & 1u);
    return (unsigned short)(u >> 16);
}

__device__ __forceinline__ int clampi(int v, int lo, int hi) {
    return v < lo ? lo : (v > hi ? hi : v);
}

// CDNA5 async copy: per-lane 16B global -> LDS, tracked with ASYNCcnt.
__device__ __forceinline__ void async_copy_b128(unsigned lds_byte_off, const void* gptr) {
    asm volatile("global_load_async_to_lds_b128 %0, %1, off"
                 :: "v"(lds_byte_off), "v"(gptr)
                 : "memory");
}
__device__ __forceinline__ void wait_async0() {
    asm volatile("s_wait_asynccnt 0x0" ::: "memory");
}

// ---------------------------------------------------------------------------
// C[M,N] = A[M,K] @ Bt[N,K]^T   A,Bt pre-converted bf16; f32 accumulate.
// Block: 128 threads = 4 waves. Tile BM=128, BN=64, BK=32, double-buffered LDS
// filled by GLOBAL_LOAD_ASYNC_TO_LDS_B128. Each wave: 32 rows x 64 cols =
// 8x v_wmma_f32_16x16x32_bf16 per K-step, B fragments reused across 2 A frags.
// ---------------------------------------------------------------------------
__global__ __launch_bounds__(128)
void gemm_bf16_wmma_async(const unsigned short* __restrict__ A,   // [M_PAD x K] bf16
                          const unsigned short* __restrict__ Bt,  // [N x K] bf16
                          float* __restrict__ C, int M, int N, int K) {
    __shared__ __align__(16) unsigned short lA[2][128 * 32];  // [row][k]  8KB/stage
    __shared__ __align__(16) unsigned short lB[2][64 * 32];   // [n][k]    4KB/stage

    const int tid  = threadIdx.x;
    const int lane = tid & 31;
    const int wave = tid >> 5;
    const int row0 = blockIdx.y * 128;
    const int n0   = blockIdx.x * 64;

    // Copy mapping: rows are 64B = 4 x 16B chunks.  A: 512 chunks (4/thread),
    // B: 256 chunks (2/thread).  chunk -> (row = id>>2, bf16 col = (id&3)*8)
    int aRow[4], aCol[4], bRow[2], bCol[2];
    const unsigned short* gA[4];
    const unsigned short* gB[2];
    unsigned offA[4], offB[2];
#pragma unroll
    for (int p = 0; p < 4; ++p) {
        int id = tid + p * 128;
        aRow[p] = id >> 2;
        aCol[p] = (id & 3) * 8;
        gA[p]   = A + (size_t)(row0 + aRow[p]) * K + aCol[p];
        offA[p] = (unsigned)(aRow[p] * 32 + aCol[p]) * 2;
    }
#pragma unroll
    for (int p = 0; p < 2; ++p) {
        int id = tid + p * 128;
        bRow[p] = id >> 2;
        bCol[p] = (id & 3) * 8;
        gB[p]   = Bt + (size_t)(n0 + bRow[p]) * K + bCol[p];
        offB[p] = (unsigned)(bRow[p] * 32 + bCol[p]) * 2;
    }

    const unsigned aBase = (unsigned)(unsigned long long)&lA[0][0];
    const unsigned bBase = (unsigned)(unsigned long long)&lB[0][0];

    // prologue: stage 0 in flight
#pragma unroll
    for (int p = 0; p < 4; ++p) async_copy_b128(aBase + offA[p], gA[p]);
#pragma unroll
    for (int p = 0; p < 2; ++p) async_copy_b128(bBase + offB[p], gB[p]);

    v8f acc[2][4];
#pragma unroll
    for (int r = 0; r < 2; ++r)
#pragma unroll
        for (int j = 0; j < 4; ++j)
#pragma unroll
            for (int i = 0; i < 8; ++i) acc[r][j][i] = 0.0f;

    const int T = K / 32;
    for (int t = 0; t < T; ++t) {
        wait_async0();        // this wave's tile t landed
        __syncthreads();      // every wave's tile t landed; tile t-1 reads done

        const int cur = t & 1;
        if (t + 1 < T) {
            const int nxt = 1 - cur;
            const int ko  = (t + 1) * 32;                 // bf16 elements
            const unsigned sbA = (unsigned)nxt * (128 * 32 * 2);
            const unsigned sbB = (unsigned)nxt * (64 * 32 * 2);
#pragma unroll
            for (int p = 0; p < 4; ++p) async_copy_b128(aBase + sbA + offA[p], gA[p] + ko);
#pragma unroll
            for (int p = 0; p < 2; ++p) async_copy_b128(bBase + sbB + offB[p], gB[p] + ko);
        }

        // A fragments: 16-bit A 16x32 layout — lane l holds row (l&15),
        // K in {kh..kh+7} and {kh+16..kh+23}, kh = (l>>4)*8.
        BfFrag a[2];
#pragma unroll
        for (int r = 0; r < 2; ++r) {
            int ar = wave * 32 + r * 16 + (lane & 15);
            int kh = (lane >> 4) * 8;
            a[r].h[0] = *(const ushort8*)&lA[cur][ar * 32 + kh];
            a[r].h[1] = *(const ushort8*)&lA[cur][ar * 32 + kh + 16];
        }
#pragma unroll
        for (int j = 0; j < 4; ++j) {
            // B fragment: lane l holds column (l&15), K in {kq..kq+7} and {kq+16..kq+23}, kq = (l>>4)*8.
            BfFrag b;
            int nc = j * 16 + (lane & 15);
            int kq = (lane >> 4) * 8;
            b.h[0] = *(const ushort8*)&lB[cur][nc * 32 + kq];
            b.h[1] = *(const ushort8*)&lB[cur][nc * 32 + kq + 16];
            acc[0][j] = __builtin_amdgcn_wmma_f32_16x16x32_bf16(
                false, a[0].v, false, b.v, (short)0, acc[0][j], false, false);
            acc[1][j] = __builtin_amdgcn_wmma_f32_16x16x32_bf16(
                false, a[1].v, false, b.v, (short)0, acc[1][j], false, false);
        }
    }

    // C/D f32 layout: lanes 0-15 -> M = base+0..7 ; lanes 16-31 -> M = base+8..15
    int colBase = n0 + (lane & 15);
#pragma unroll
    for (int r = 0; r < 2; ++r) {
        int mBase = row0 + wave * 32 + r * 16 + (lane >> 4) * 8;
#pragma unroll
        for (int j = 0; j < 4; ++j) {
            int col = colBase + j * 16;
#pragma unroll
            for (int i = 0; i < 8; ++i) {
                int m = mBase + i;
                if (m < M) C[(size_t)m * N + col] = acc[r][j][i];
            }
        }
    }
}

// ---------------------------------------------------------------------------
// Conversion / transpose pre-passes
// ---------------------------------------------------------------------------
__global__ void cvt_bf16_k(const float* __restrict__ in, unsigned short* __restrict__ out,
                           long long total) {
    long long i = (long long)blockIdx.x * blockDim.x + threadIdx.x;
    if (i < total) out[i] = f2bf(in[i]);
}

// Wt[n*K + k] = bf16(W[k*N + n])   (768x768, coalesced read)
__global__ void wtrans_k(const float* __restrict__ W, unsigned short* __restrict__ Wt) {
    int i = blockIdx.x * blockDim.x + threadIdx.x;
    if (i < D_MODEL * D_MODEL) {
        int k = i / D_MODEL;
        int n = i % D_MODEL;
        Wt[(size_t)n * D_MODEL + k] = f2bf(W[i]);
    }
}

// ---------------------------------------------------------------------------
// Degree / normalization helpers
// ---------------------------------------------------------------------------
__global__ void fill1_k(float* __restrict__ p, int n) {
    int i = blockIdx.x * blockDim.x + threadIdx.x;
    if (i < n) p[i] = 1.0f;
}

__global__ void deg_scatter_k(float* __restrict__ deg,
                              const int* __restrict__ edge, int nE, int nN) {
    int i = blockIdx.x * blockDim.x + threadIdx.x;
    if (i < nE) {
        int d = clampi(edge[nE + i], 0, nN - 1);
        atomicAdd(&deg[d], 1.0f);
    }
}

__global__ void rsqrt_k(float* __restrict__ p, int n) {
    int i = blockIdx.x * blockDim.x + threadIdx.x;
    if (i < n) p[i] = rsqrtf(p[i]);
}

// ---------------------------------------------------------------------------
// agg = h * dinv^2 + b   (self-loop contribution + bias, base for scatter)
// ---------------------------------------------------------------------------
__global__ void agg_init_k(const float* __restrict__ H, const float* __restrict__ dinv,
                           const float* __restrict__ bias, float* __restrict__ AGG,
                           long long total) {
    long long i = (long long)blockIdx.x * blockDim.x + threadIdx.x;
    if (i < total) {
        int node = (int)(i / D_MODEL);
        int f    = (int)(i % D_MODEL);
        float di = dinv[node];
        AGG[i] = H[i] * di * di + bias[f];
    }
}

// one block per edge; 256 threads cover 768 features (3 each)
__global__ __launch_bounds__(256)
void edge_scatter_k(const float* __restrict__ H, const float* __restrict__ dinv,
                    const int* __restrict__ edge, float* __restrict__ AGG) {
    int e = blockIdx.x;
    int s = clampi(edge[e], 0, N_NODES - 1);
    int d = clampi(edge[N_EDGES + e], 0, N_NODES - 1);
    float norm = dinv[s] * dinv[d];
    const float* hs = H + (size_t)s * D_MODEL;
    float* ad = AGG + (size_t)d * D_MODEL;
#pragma unroll
    for (int j = 0; j < 3; ++j) {
        int f = threadIdx.x + j * 256;
        atomicAdd(&ad[f], hs[f] * norm);
    }
}

// ReLU: write f32 result and bf16 copy (next layer's WMMA operand)
__global__ void relu_store_k(float* __restrict__ Q, unsigned short* __restrict__ Hh,
                             long long total) {
    long long i = (long long)blockIdx.x * blockDim.x + threadIdx.x;
    if (i < total) {
        float r = fmaxf(Q[i], 0.0f);
        Q[i]  = r;
        Hh[i] = f2bf(r);
    }
}

__global__ void residual_k(const float* __restrict__ H, const float* __restrict__ X,
                           float* __restrict__ OUT, long long total) {
    long long i = (long long)blockIdx.x * blockDim.x + threadIdx.x;
    if (i < total) OUT[i] = H[i] + X[i];
}

__global__ void zero_k(float* __restrict__ p, int n) {
    int i = blockIdx.x * blockDim.x + threadIdx.x;
    if (i < n) p[i] = 0.0f;
}

// column sums of h: grid (3, rowChunks); 128 rows per chunk, 1 atomic per thread
__global__ __launch_bounds__(256)
void colsum_k(const float* __restrict__ H, float* __restrict__ colsum, int M) {
    int col = blockIdx.x * 256 + threadIdx.x;           // 0..767
    int r0  = blockIdx.y * 128;
    int r1  = min(r0 + 128, M);
    float s = 0.0f;
    for (int r = r0; r < r1; ++r) s += H[(size_t)r * D_MODEL + col];
    atomicAdd(&colsum[col], s);
}

// out[n] = mean_h . Wl[:,n] + bl[n]   (mean-pool commuted through the linear layer)
__global__ __launch_bounds__(256)
void final_linear_k(const float* __restrict__ colsum, const float* __restrict__ Wl,
                    const float* __restrict__ bl, float* __restrict__ out) {
    int n = blockIdx.x * 256 + threadIdx.x;             // 0..767
    const float invN = 1.0f / (float)N_NODES;
    float acc = bl[n];
    for (int k = 0; k < D_MODEL; ++k)
        acc += (colsum[k] * invN) * Wl[(size_t)k * D_MODEL + n];
    out[n] = acc;
}

// ---------------------------------------------------------------------------
extern "C" void kernel_launch(void* const* d_in, const int* in_sizes, int n_in,
                              void* d_out, int out_size, void* d_ws, size_t ws_size,
                              hipStream_t stream) {
    const float* X    = (const float*)d_in[0];
    const int*   edge = (const int*)d_in[1];   // [2,E] flat: [0..E)=src, [E..2E)=dst
    const float* W1 = (const float*)d_in[2];
    const float* b1 = (const float*)d_in[3];
    const float* W2 = (const float*)d_in[4];
    const float* b2 = (const float*)d_in[5];
    const float* W3 = (const float*)d_in[6];
    const float* b3 = (const float*)d_in[7];
    const float* Wl = (const float*)d_in[8];
    const float* bl = (const float*)d_in[9];

    const long long ND = (long long)N_NODES * D_MODEL;
    float* outH = (float*)d_out;          // h  [50000,768]
    float* outP = outH + ND;              // out[1,768]

    // workspace: dinv | colsum | P | Q | Hh(bf16, padded rows) | Wt(bf16)
    float* dinv   = (float*)d_ws;
    float* colsum = dinv + 50176;
    float* P      = colsum + 1024;
    float* Q      = P + ND;
    unsigned short* Hh = (unsigned short*)(Q + ND);        // M_PAD x 768 bf16
    unsigned short* Wt = Hh + (size_t)M_PAD * D_MODEL;     // 768 x 768 bf16

    const int ndBlocks = (int)((ND + 255) / 256);
    const int nBlocks  = (N_NODES + 255) / 256;
    const int eBlocks  = (N_EDGES + 255) / 256;
    const int wBlocks  = (D_MODEL * D_MODEL + 255) / 256;

    // degrees with self-loops -> dinv = rsqrt(1 + indeg)
    fill1_k<<<nBlocks, 256, 0, stream>>>(dinv, N_NODES);
    deg_scatter_k<<<eBlocks, 256, 0, stream>>>(dinv, edge, N_EDGES, N_NODES);
    rsqrt_k<<<nBlocks, 256, 0, stream>>>(dinv, N_NODES);

    // layer-1 input in bf16
    cvt_bf16_k<<<ndBlocks, 256, 0, stream>>>(X, Hh, ND);

    dim3 gemmGrid(D_MODEL / 64, M_PAD / 128);             // 12 x 391
    const float* Ws[3] = {W1, W2, W3};
    const float* bs[3] = {b1, b2, b3};
    for (int l = 0; l < 3; ++l) {
        wtrans_k<<<wBlocks, 256, 0, stream>>>(Ws[l], Wt);
        gemm_bf16_wmma_async<<<gemmGrid, 128, 0, stream>>>(Hh, Wt, P,
                                                           N_NODES, D_MODEL, D_MODEL);
        agg_init_k<<<ndBlocks, 256, 0, stream>>>(P, dinv, bs[l], Q, ND);
        edge_scatter_k<<<N_EDGES, 256, 0, stream>>>(P, dinv, edge, Q);
        relu_store_k<<<ndBlocks, 256, 0, stream>>>(Q, Hh, ND);   // Hh = next input
    }

    // h = h3 + x  (written straight to output)
    residual_k<<<ndBlocks, 256, 0, stream>>>(Q, X, outH, ND);

    // out = mean(h) @ Wl + bl  (mean commutes with the linear layer)
    zero_k<<<4, 256, 0, stream>>>(colsum, 1024);
    dim3 csGrid(3, (N_NODES + 127) / 128);
    colsum_k<<<csGrid, 256, 0, stream>>>(outH, colsum, N_NODES);
    final_linear_k<<<3, 256, 0, stream>>>(colsum, Wl, bl, outP);
}
